// ScaledDotProductAttention_1924145349283
// MI455X (gfx1250) — matrix operand order repaired and checked by reading, not compiled, not run on hardware
//
#include <hip/hip_runtime.h>
#include <hip/hip_bf16.h>

typedef __attribute__((ext_vector_type(16))) __bf16 v16bf;
typedef __attribute__((ext_vector_type(8)))  float  v8f;

#define S_LEN 2048
#define D_LEN 64
#define NBLK  64                 // keys processed per iteration
#define NITER (S_LEN / NBLK)
#define SCALE 0.125f             // 64^-0.5

// LDS strides (bf16 elements): 72*2 = 144B rows (16B aligned, good bank spread)
#define KT_STRIDE 72             // K^T tile: [key 0..63][d 0..63]
#define VT_STRIDE 72             // V^T tile: [d 0..63][key 0..63]
#define P_STRIDE  72             // per-wave P scratch: [m 0..15][key 0..63]

union Frag16 { uint4 u[2]; v16bf f; };

// 16-lane (DPP row) max butterfly: xor1, xor2, half-mirror, mirror. Pure VALU.
__device__ __forceinline__ float rowmax16(float x) {
    int v;
    v = __builtin_amdgcn_update_dpp(0, __float_as_int(x), 0xB1,  0xF, 0xF, true); // quad xor1
    x = fmaxf(x, __int_as_float(v));
    v = __builtin_amdgcn_update_dpp(0, __float_as_int(x), 0x4E,  0xF, 0xF, true); // quad xor2
    x = fmaxf(x, __int_as_float(v));
    v = __builtin_amdgcn_update_dpp(0, __float_as_int(x), 0x141, 0xF, 0xF, true); // row_half_mirror
    x = fmaxf(x, __int_as_float(v));
    v = __builtin_amdgcn_update_dpp(0, __float_as_int(x), 0x140, 0xF, 0xF, true); // row_mirror
    x = fmaxf(x, __int_as_float(v));
    return x;
}

__global__ __launch_bounds__(256, 1)
void attn_fused(const float* __restrict__ q, const float* __restrict__ k,
                const float* __restrict__ v, const float* __restrict__ prev,
                float* __restrict__ out, float* __restrict__ scores)
{
    __shared__ __align__(16) __bf16 ldsK[64 * KT_STRIDE];
    __shared__ __align__(16) __bf16 ldsV[64 * VT_STRIDE];
    __shared__ __align__(16) __bf16 ldsP[8][16 * P_STRIDE];

    const int tid  = threadIdx.x;
    const int wave = tid >> 5;
    const int lane = tid & 31;
    const int h    = lane >> 4;      // lane half (WMMA layout)
    const int ln   = lane & 15;
    const int bh   = blockIdx.x;                       // 0..31 (B*H)
    const int qbase = (blockIdx.y * 8 + wave) * 16;    // this wave's 16 query rows

    const float* qb = q    + (size_t)bh * S_LEN * D_LEN;
    const float* kb = k    + (size_t)bh * D_LEN * S_LEN;   // k is [D][S]
    const float* vb = v    + (size_t)bh * S_LEN * D_LEN;
    const float* pb = prev + (size_t)bh * S_LEN * S_LEN;
    float*       sb = scores + (size_t)bh * S_LEN * S_LEN;
    float*       ob = out    + (size_t)bh * S_LEN * D_LEN;

    // ---- Q tile 16x64 -> two bf16 A-fragments (K = d 0..31 and 32..63) ----
    // A operand as packed here (bf16 16x32, contiguous k order): lane: M=ln; element i: K = 16*h + i
    v16bf qa[2];
    {
        const float* qrow = qb + (size_t)(qbase + ln) * D_LEN;
        #pragma unroll
        for (int op = 0; op < 2; ++op) {
            #pragma unroll
            for (int i = 0; i < 16; ++i) {
                int d = op * 32 + 16 * h + i;
                qa[op][i] = (__bf16)qrow[d];
            }
        }
    }

    // constant all-ones B fragment: folds the softmax row-sum into a WMMA
    v16bf ones;
    #pragma unroll
    for (int i = 0; i < 16; ++i) ones[i] = (__bf16)1.0f;

    // accumulators: O 16x64 as 4 fragments + running row-sum l as a 5th fragment
    v8f of[4], lf;
    #pragma unroll
    for (int f = 0; f < 4; ++f)
        #pragma unroll
        for (int r = 0; r < 8; ++r) of[f][r] = 0.0f;
    #pragma unroll
    for (int r = 0; r < 8; ++r) lf[r] = 0.0f;

    float m_[8];
    #pragma unroll
    for (int r = 0; r < 8; ++r) m_[r] = -__builtin_inff();

    #pragma unroll 1
    for (int nb = 0; nb < NITER; ++nb) {
        const int keybase = nb * NBLK;

        __syncthreads();   // previous iteration's LDS reads done

        // ---- cooperative fill: K^T tile (LDS [key][d]), coalesced along keys ----
        {
            const int keyrel = tid & 63, dg = tid >> 6;   // dg: 0..3 (16 d each)
            union { __bf16 hb[16]; uint4 u[2]; } pk;
            #pragma unroll
            for (int j = 0; j < 16; ++j)
                pk.hb[j] = (__bf16)kb[(size_t)(dg * 16 + j) * S_LEN + keybase + keyrel];
            uint4* dst = (uint4*)&ldsK[keyrel * KT_STRIDE + dg * 16];
            dst[0] = pk.u[0];
            dst[1] = pk.u[1];
        }
        // ---- cooperative fill: V^T tile (LDS [d][key]), key-pair packed b32 ----
        {
            const int kp = tid >> 3, dg = tid & 7;        // kp: 0..31 key pairs, dg: d/8
            const float* vr0 = vb + (size_t)(keybase + 2 * kp) * D_LEN + dg * 8;
            float4 a0 = *(const float4*)vr0;
            float4 a1 = *(const float4*)(vr0 + 4);
            float4 b0 = *(const float4*)(vr0 + D_LEN);
            float4 b1 = *(const float4*)(vr0 + D_LEN + 4);
            float v0[8] = {a0.x, a0.y, a0.z, a0.w, a1.x, a1.y, a1.z, a1.w};
            float v1[8] = {b0.x, b0.y, b0.z, b0.w, b1.x, b1.y, b1.z, b1.w};
            #pragma unroll
            for (int j = 0; j < 8; ++j) {
                union { __bf16 hb[2]; unsigned u; } pp;
                pp.hb[0] = (__bf16)v0[j];
                pp.hb[1] = (__bf16)v1[j];
                *(unsigned*)&ldsV[(dg * 8 + j) * VT_STRIDE + 2 * kp] = pp.u;
            }
        }
        __syncthreads();

        // ---- GEMM1: S[16 x 64keys] = Q(16x64) * K(64x64) ----
        // Key interleave by 4: tile t, col N=ln -> key 4*ln + t  => b128 global I/O
        v8f s[4];
        #pragma unroll
        for (int t = 0; t < 4; ++t)
            #pragma unroll
            for (int r = 0; r < 8; ++r) s[t][r] = 0.0f;
        #pragma unroll
        for (int dop = 0; dop < 2; ++dop) {
            #pragma unroll
            for (int t = 0; t < 4; ++t) {
                Frag16 bt;
                const __bf16* kp_ = &ldsK[(4 * ln + t) * KT_STRIDE + dop * 32 + 16 * h];
                bt.u[0] = *(const uint4*)kp_;
                bt.u[1] = *(const uint4*)(kp_ + 8);
                s[t] = __builtin_amdgcn_wmma_f32_16x16x32_bf16(false, qa[dop], false, bt.f,
                                                               (short)0, s[t], false, false);
            }
        }

        // ---- scale + prev, write attn_scores (float4: lane = keys 4ln..4ln+3) ----
        const size_t rowbase = (size_t)(qbase + 8 * h) * S_LEN + keybase + 4 * ln;
        float4 pv8[8];
        #pragma unroll
        for (int r = 0; r < 8; ++r)                     // batch loads into one clause
            pv8[r] = *(const float4*)(pb + rowbase + (size_t)r * S_LEN);
        #pragma unroll
        for (int r = 0; r < 8; ++r)
            __builtin_prefetch((const void*)(pb + rowbase + (size_t)r * S_LEN + NBLK), 0, 3);
        #pragma unroll
        for (int r = 0; r < 8; ++r) {
            const size_t idx = rowbase + (size_t)r * S_LEN;
            float x0 = s[0][r] * SCALE + pv8[r].x;
            float x1 = s[1][r] * SCALE + pv8[r].y;
            float x2 = s[2][r] * SCALE + pv8[r].z;
            float x3 = s[3][r] * SCALE + pv8[r].w;
            *(float4*)(sb + idx) = make_float4(x0, x1, x2, x3);
            s[0][r] = x0;  s[1][r] = x1;  s[2][r] = x2;  s[3][r] = x3;
        }

        // ---- online softmax: DPP row-max, exp, rescale, stash P in LDS ----
        #pragma unroll
        for (int r = 0; r < 8; ++r) {
            const float tm   = rowmax16(fmaxf(fmaxf(s[0][r], s[1][r]),
                                              fmaxf(s[2][r], s[3][r])));
            const float mnew  = fmaxf(m_[r], tm);
            const float alpha = __expf(m_[r] - mnew);
            m_[r] = mnew;
            of[0][r] *= alpha; of[1][r] *= alpha; of[2][r] *= alpha; of[3][r] *= alpha;
            lf[r]    *= alpha;
            union { __bf16 hb[4]; uint2 u; } pp;
            #pragma unroll
            for (int t = 0; t < 4; ++t)
                pp.hb[t] = (__bf16)__expf(s[t][r] - mnew);
            // packed quad store: keys 4ln..4ln+3 of row M = r+8h (one b64)
            *(uint2*)&ldsP[wave][(r + 8 * h) * P_STRIDE + 4 * ln] = pp.u;
        }

        // wave-internal LDS ordering: DS ops are in-order per wave; block compiler motion
        asm volatile("s_wait_dscnt 0x0" ::: "memory");
        __builtin_amdgcn_wave_barrier();

        // ---- reload P as two A-fragments: M=ln, element i: key=kop*32+16h+i ----
        Frag16 pa[2];
        #pragma unroll
        for (int kop = 0; kop < 2; ++kop) {
            pa[kop].u[0] = *(const uint4*)&ldsP[wave][ln * P_STRIDE + kop * 32 + 16 * h];
            pa[kop].u[1] = *(const uint4*)&ldsP[wave][ln * P_STRIDE + kop * 32 + 16 * h + 8];
        }

        // ---- GEMM2: O(16x64) += P(16x64) * V(64x64); + row-sum via ones-WMMA ----
        #pragma unroll
        for (int kop = 0; kop < 2; ++kop) {
            #pragma unroll
            for (int f = 0; f < 4; ++f) {
                Frag16 vf;
                const __bf16* vp = &ldsV[(f * 16 + ln) * VT_STRIDE + kop * 32 + 16 * h];
                vf.u[0] = *(const uint4*)vp;
                vf.u[1] = *(const uint4*)(vp + 8);
                of[f] = __builtin_amdgcn_wmma_f32_16x16x32_bf16(false, pa[kop].f, false, vf.f,
                                                                (short)0, of[f], false, false);
            }
            lf = __builtin_amdgcn_wmma_f32_16x16x32_bf16(false, pa[kop].f, false, ones,
                                                         (short)0, lf, false, false);
        }
    }

    // ---- epilogue: O /= l, write coalesced ----
    #pragma unroll
    for (int r = 0; r < 8; ++r) {
        const float inv = 1.0f / lf[r];
        const size_t row = (size_t)(qbase + r + 8 * h);
        #pragma unroll
        for (int f = 0; f < 4; ++f)
            ob[row * D_LEN + f * 16 + ln] = of[f][r] * inv;
    }
}

extern "C" void kernel_launch(void* const* d_in, const int* in_sizes, int n_in,
                              void* d_out, int out_size, void* d_ws, size_t ws_size,
                              hipStream_t stream) {
    (void)in_sizes; (void)n_in; (void)out_size; (void)d_ws; (void)ws_size;
    const float* q    = (const float*)d_in[0];
    const float* k    = (const float*)d_in[1];
    const float* v    = (const float*)d_in[2];
    const float* prev = (const float*)d_in[3];

    float* out    = (float*)d_out;                                  // [B,H,S,D]
    float* scores = out + (size_t)2 * 16 * S_LEN * D_LEN;           // [B,H,S,S]

    dim3 grid(32, S_LEN / (16 * 8));   // (B*H, query-chunks of 128 rows)
    dim3 block(256);                   // 8 waves of 32
    attn_fused<<<grid, block, 0, stream>>>(q, k, v, prev, out, scores);
}
